// CausalSelfAttention_2319282340480
// MI455X (gfx1250) — matrix operand order repaired and checked by reading, not compiled, not run on hardware
//
#include <hip/hip_runtime.h>
#include <hip/hip_bf16.h>

#define BB 16
#define SS 2048
#define DD 512

typedef __attribute__((ext_vector_type(16))) __bf16 v16bf;
typedef __attribute__((ext_vector_type(8)))  __bf16 v8bf;
typedef __attribute__((ext_vector_type(8)))  float  v8f;

union V16U { v16bf v; v8bf h[2]; };

static __device__ __forceinline__ v8f wmma_bf16(v16bf a, v16bf b, v8f c) {
  // 8 args: (neg_a, A, neg_b, B, c_mod, C, reuse_a, reuse_b)
  return __builtin_amdgcn_wmma_f32_16x16x32_bf16(false, a, false, b, (short)0, c,
                                                 false, false);
}

// ---------------------------------------------------------------------------
// Kernel 1: convert Wq/Wk/Wv (f32 [D][D]) to bf16, packed [3][D][D]
// ---------------------------------------------------------------------------
__global__ void cvt_w_kernel(const float* __restrict__ wq,
                             const float* __restrict__ wk,
                             const float* __restrict__ wv,
                             __bf16* __restrict__ wbf) {
  const int i = blockIdx.x * blockDim.x + threadIdx.x;
  const int n = DD * DD;
  if (i < n) {
    wbf[i]         = (__bf16)wq[i];
    wbf[i + n]     = (__bf16)wk[i];
    wbf[i + 2 * n] = (__bf16)wv[i];
  }
}

// ---------------------------------------------------------------------------
// Kernel 2: QKV projection.  One block per 16-row x tile; 8 waves split the
// 96 (matrix, n-tile) outputs.  B-fragment loads are double-buffered so the
// load for K-step kk+1 issues before the WMMA of step kk.
// Q,K stored bf16 row-major; V stored transposed Vt[b][d][s].
// ---------------------------------------------------------------------------
__global__ void __launch_bounds__(256)
qkv_kernel(const float* __restrict__ x,
           const __bf16* __restrict__ wbf,
           const float* __restrict__ bq,
           const float* __restrict__ bk,
           const float* __restrict__ bv,
           __bf16* __restrict__ qbf,
           __bf16* __restrict__ kbf,
           __bf16* __restrict__ vt) {
  __shared__ __bf16 sX[16][DD];  // 16 KB

  const int tid = threadIdx.x;
  const int b   = blockIdx.x >> 7;
  const int s0  = (blockIdx.x & 127) << 4;

  // stage x tile into LDS as bf16
  const float* xrow = x + ((size_t)(b * SS + s0)) * DD;
  for (int i = tid; i < 16 * DD; i += 256) {
    const int r = i >> 9, c = i & (DD - 1);
    sX[r][c] = (__bf16)xrow[(size_t)r * DD + c];
  }
  __syncthreads();

  const int wid  = tid >> 5;
  const int lane = tid & 31;
  const int lo   = lane & 15;
  const int hi   = lane >> 4;

  // hoist all 16 x A-fragments into registers (loop-invariant across tiles)
  V16U aX[16];
  #pragma unroll
  for (int kk = 0; kk < 16; ++kk) {
    aX[kk].h[0] = *(const v8bf*)&sX[lo][kk * 32 + hi * 16];
    aX[kk].h[1] = *(const v8bf*)&sX[lo][kk * 32 + hi * 16 + 8];
  }

  for (int t = wid; t < 96; t += 8) {
    const int m  = t >> 5;          // 0=Q, 1=K, 2=V
    const int nb = (t & 31) << 4;   // output column base
    const __bf16* wrow = wbf + (size_t)m * DD * DD + (size_t)(nb + lo) * DD +
                         hi * 16;

    v8f c = {};
    v16bf bcur = *(const v16bf*)(wrow);
    #pragma unroll
    for (int kk = 0; kk < 16; ++kk) {
      v16bf bnxt = {};
      if (kk < 15) bnxt = *(const v16bf*)(wrow + (kk + 1) * 32);
      c = wmma_bf16(aX[kk].v, bcur, c);
      bcur = bnxt;
    }

    const int col = nb + lo;
    const float bias = (m == 0 ? bq[col] : (m == 1 ? bk[col] : bv[col]));
    #pragma unroll
    for (int v = 0; v < 8; ++v) c[v] += bias;

    if (m < 2) {
      __bf16* dst = (m == 0 ? qbf : kbf);
      #pragma unroll
      for (int v = 0; v < 8; ++v) {
        const int s = s0 + v + hi * 8;                    // C row = v + 8*hi
        dst[((size_t)(b * SS + s)) * DD + col] = (__bf16)c[v];
      }
    } else {
      v8bf pk;
      #pragma unroll
      for (int v = 0; v < 8; ++v) pk[v] = (__bf16)c[v];
      *(v8bf*)(vt + ((size_t)(b * DD + col)) * SS + s0 + hi * 8) = pk;
    }
  }
}

// ---------------------------------------------------------------------------
// Kernel 3: flash attention.  One block per (b, 16-query tile); 8 waves split
// D into 64-wide slices.  Scores computed cooperatively (d-split partials
// reduced via LDS), wave 0 does masking + online softmax, all waves do P@V.
// K/V B-fragments for a k-block are batch-loaded before their WMMA chains,
// and the next k-block is prefetched (global_prefetch) while computing.
// ---------------------------------------------------------------------------
__global__ void __launch_bounds__(256)
attn_kernel(const __bf16* __restrict__ qbf,
            const __bf16* __restrict__ kbf,
            const __bf16* __restrict__ vt,
            const unsigned char* __restrict__ pad,
            float* __restrict__ out) {
  __shared__ float  sPart[2][8][256];  // 16 KB partial score tiles
  __shared__ float  sS[16][32];        // masked scores, row-major
  __shared__ __bf16 sP[16][32];        // probabilities (bf16)
  __shared__ float  sF[16];            // per-row O rescale factor
  __shared__ float  sL[16];            // per-row running sum

  const int tid  = threadIdx.x;
  const int w    = tid >> 5;
  const int lane = tid & 31;
  const int lo   = lane & 15;
  const int hi   = lane >> 4;
  const int b    = blockIdx.x >> 7;
  const int q0   = (blockIdx.x & 127) << 4;
  const int dw0  = w << 6;             // this wave's 64-wide d slice

  const float scale = 0.04419417382415922f;  // 1/sqrt(512)

  // Q A-fragments for this wave's d slice (loaded once)
  V16U aQ[2];
  {
    const __bf16* qrow = qbf + ((size_t)(b * SS + q0 + lo)) * DD + dw0;
    #pragma unroll
    for (int f = 0; f < 2; ++f) {
      aQ[f].h[0] = *(const v8bf*)(qrow + f * 32 + hi * 16);
      aQ[f].h[1] = *(const v8bf*)(qrow + f * 32 + hi * 16 + 8);
    }
  }

  v8f o[4] = {};
  float m_run = -1e30f, l_run = 0.f;   // live on wave 0, lanes 0..15

  const int nkb = (q0 + 47) >> 5;      // causal: keys 0 .. q0+15
  for (int kb = 0; kb < nkb; ++kb) {
    const int kbase = kb << 5;

    // prefetch next k-block's K and Vt rows into near caches
    if (kb + 1 < nkb) {
      const int kn = kbase + 32;
      __builtin_prefetch(
          kbf + ((size_t)(b * SS + kn + lo)) * DD + dw0 + hi * 16, 0, 0);
      __builtin_prefetch(
          vt + ((size_t)(b * DD + dw0 + lo)) * SS + kn + hi * 16, 0, 0);
    }

    // ---- Phase 1: partial scores over this wave's d slice ----
    {
      v16bf bK[2][2];
      #pragma unroll
      for (int t = 0; t < 2; ++t) {
        const __bf16* krow =
            kbf + ((size_t)(b * SS + kbase + t * 16 + lo)) * DD + dw0 +
            hi * 16;
        bK[t][0] = *(const v16bf*)(krow);
        bK[t][1] = *(const v16bf*)(krow + 32);
      }
      #pragma unroll
      for (int t = 0; t < 2; ++t) {
        v8f p = {};
        p = wmma_bf16(aQ[0].v, bK[t][0], p);
        p = wmma_bf16(aQ[1].v, bK[t][1], p);
        *(v8f*)&sPart[t][w][lane * 8] = p;
      }
    }
    __syncthreads();

    // ---- Phase 2: wave 0 reduces partials, masks, online softmax ----
    if (w == 0) {
      #pragma unroll
      for (int t = 0; t < 2; ++t) {
        v8f red = {};
        #pragma unroll
        for (int ww = 0; ww < 8; ++ww) {
          v8f part = *(const v8f*)&sPart[t][ww][lane * 8];
          #pragma unroll
          for (int v = 0; v < 8; ++v) red[v] += part[v];
        }
        #pragma unroll
        for (int v = 0; v < 8; ++v) {
          const int r  = v + hi * 8;           // C row
          const int kg = kbase + t * 16 + lo;  // global key index
          float s = red[v] * scale;
          if (kg > q0 + r || pad[b * SS + kg]) s = -1e30f;
          sS[r][t * 16 + lo] = s;
        }
      }
      if (lane < 16) {
        const int r = lane;
        float sc[32];
        float mnew = m_run;
        #pragma unroll
        for (int c = 0; c < 32; ++c) {
          sc[c] = sS[r][c];
          mnew  = fmaxf(mnew, sc[c]);
        }
        float sum = 0.f;
        #pragma unroll
        for (int c = 0; c < 32; ++c) {
          const float pv = __expf(sc[c] - mnew);
          sum += pv;
          sP[r][c] = (__bf16)pv;
        }
        const float fac = __expf(m_run - mnew);
        l_run = l_run * fac + sum;
        m_run = mnew;
        sF[r] = fac;
        sL[r] = l_run;
      }
    }
    __syncthreads();

    // ---- Phase 3: rescale O, P@V for this wave's n slice ----
    {
      v16bf bV[4];
      #pragma unroll
      for (int nt = 0; nt < 4; ++nt) {
        const int n = dw0 + nt * 16 + lo;
        bV[nt] = *(const v16bf*)(vt + ((size_t)(b * DD + n)) * SS + kbase +
                                 hi * 16);
      }
      v8f fr = *(const v8f*)&sF[hi * 8];
      V16U aP;
      aP.h[0] = *(const v8bf*)&sP[lo][hi * 16];
      aP.h[1] = *(const v8bf*)&sP[lo][hi * 16 + 8];
      #pragma unroll
      for (int nt = 0; nt < 4; ++nt) {
        #pragma unroll
        for (int v = 0; v < 8; ++v) o[nt][v] *= fr[v];
        o[nt] = wmma_bf16(aP.v, bV[nt], o[nt]);
      }
    }
  }

  // ---- Epilogue: divide by running sum, zero padded query rows ----
  v8f lv = *(const v8f*)&sL[hi * 8];
  #pragma unroll
  for (int nt = 0; nt < 4; ++nt) {
    const int n = dw0 + nt * 16 + lo;
    #pragma unroll
    for (int v = 0; v < 8; ++v) {
      const int q = q0 + v + hi * 8;
      float val = (lv[v] > 0.f) ? o[nt][v] / lv[v] : 0.f;
      if (pad[b * SS + q]) val = 0.f;
      out[((size_t)(b * SS + q)) * DD + n] = val;
    }
  }
}

// ---------------------------------------------------------------------------
extern "C" void kernel_launch(void* const* d_in, const int* in_sizes, int n_in,
                              void* d_out, int out_size, void* d_ws,
                              size_t ws_size, hipStream_t stream) {
  (void)in_sizes; (void)n_in; (void)out_size; (void)ws_size;

  const float* x  = (const float*)d_in[0];
  const float* Wq = (const float*)d_in[1];
  const float* bq = (const float*)d_in[2];
  const float* Wk = (const float*)d_in[3];
  const float* bk = (const float*)d_in[4];
  const float* Wv = (const float*)d_in[5];
  const float* bv = (const float*)d_in[6];
  const unsigned char* pad = (const unsigned char*)d_in[7];
  float* out = (float*)d_out;

  // workspace layout (bf16): [Wbf 3*D*D][Q B*S*D][K B*S*D][Vt B*D*S] ~98 MB
  __bf16* wbf = (__bf16*)d_ws;
  __bf16* qbf = wbf + (size_t)3 * DD * DD;
  __bf16* kbf = qbf + (size_t)BB * SS * DD;
  __bf16* vtp = kbf + (size_t)BB * SS * DD;

  cvt_w_kernel<<<(DD * DD + 255) / 256, 256, 0, stream>>>(Wq, Wk, Wv, wbf);
  qkv_kernel<<<BB * (SS / 16), 256, 0, stream>>>(x, wbf, bq, bk, bv,
                                                 qbf, kbf, vtp);
  attn_kernel<<<BB * (SS / 16), 256, 0, stream>>>(qbf, kbf, vtp, pad, out);
}
